// RecurrentNetwork_43173011259937
// MI455X (gfx1250) — matrix operand order repaired and checked by reading, not compiled, not run on hardware
//
#include <hip/hip_runtime.h>
#include <hip/hip_bf16.h>

typedef __bf16 bf16;
typedef __attribute__((ext_vector_type(16))) __bf16 v16bf;
typedef __attribute__((ext_vector_type(8)))  __bf16 v8bf;
typedef __attribute__((ext_vector_type(8)))  float  v8f;

union FragBF { v16bf v; v8bf h[2]; };

__device__ __forceinline__ v8f wmma_bf16(v16bf a, v16bf b, v8f c) {
  // D = A(16x32) * B(32x16) + C, fp32 accumulate
  return __builtin_amdgcn_wmma_f32_16x16x32_bf16(
      /*neg_a=*/false, a, /*neg_b=*/false, b,
      /*c_mod=*/(short)0, c, /*reuse_a=*/false, /*reuse_b=*/false);
}

// ---------------------------------------------------------------- convert
__global__ void cvt_f32_bf16(const float* __restrict__ src, bf16* __restrict__ dst, int n) {
  int i = blockIdx.x * blockDim.x + threadIdx.x;
  int stride = gridDim.x * blockDim.x;
  for (; i < n; i += stride) dst[i] = (bf16)src[i];
}

// ---------------------------------------------------------------- input projection
// Y[(t*32 + b)*N + j] = sum_k X[(b*T+t)*K + k] * W[j*K + k] + b_ih[j] + b_hh[j]
// X bf16 [M=B*T][K] row-major, W bf16 [N][K] row-major (already B-fragment friendly).
// Register blocking: each wave owns 1 M-tile x 4 N-tiles -> 4 WMMAs per A fragment.
__global__ void proj_kernel(const bf16* __restrict__ X, const bf16* __restrict__ W,
                            const float* __restrict__ b_ih, const float* __restrict__ b_hh,
                            float* __restrict__ Y, int M, int K, int N, int T)
{
  const int lane = threadIdx.x & 31;
  const int lm = lane & 15, hi = lane >> 4;
  const int wave   = (blockIdx.x * blockDim.x + threadIdx.x) >> 5;
  const int nwaves = (gridDim.x * blockDim.x) >> 5;
  const int mtiles  = M >> 4;
  const int ngroups = N >> 6;                 // groups of 4 N-tiles (64 columns)
  const int groups  = mtiles * ngroups;

  for (int g = wave; g < groups; g += nwaves) {
    const int mt = g / ngroups, ng = g % ngroups;
    const int m0 = mt << 4, n0 = ng << 6;
    const bf16* arow = X + (size_t)(m0 + lm) * K;
    const bf16* brow = W + (size_t)(n0 + lm) * K;
    v8f acc[4] = {};
    for (int k0 = 0; k0 < K; k0 += 32) {
      FragBF a;
      a.h[0] = *(const v8bf*)(arow + k0 + hi * 8);
      a.h[1] = *(const v8bf*)(arow + k0 + hi * 8 + 16);
      const bf16* bk = brow + k0 + hi * 8;
#pragma unroll
      for (int s = 0; s < 4; ++s) {
        FragBF b;
        b.h[0] = *(const v8bf*)(bk + (size_t)s * 16 * K);
        b.h[1] = *(const v8bf*)(bk + (size_t)s * 16 * K + 16);
        acc[s] = wmma_bf16(a.v, b.v, acc[s]);
      }
    }
#pragma unroll
    for (int s = 0; s < 4; ++s) {
      const int j = n0 + s * 16 + lm;
      const float bias = b_ih[j] + b_hh[j];
#pragma unroll
      for (int r = 0; r < 8; ++r) {
        const int m = m0 + hi * 8 + r;          // C/D layout: m = r + 8*hi
        const int bb = m / T, tt = m % T;       // X rows are m = b*T + t
        Y[((size_t)tt * 32 + bb) * N + j] = acc[s][r] + bias;
      }
    }
  }
}

// ---------------------------------------------------------------- recurrence
// h_t = tanh(xp_t + h_{t-1} @ Whh^T). Whh slice LDS-resident.
// NWG>1: h double-buffered in global, staged into LDS once per step; ticket barrier.
// NWG==1: h double-buffered entirely in LDS (no global h traffic on the critical path).
template<int H, int NWG, bool LAST>
__global__ __launch_bounds__(512)
void rnn_step_kernel(const bf16* __restrict__ Whh,  // [H][H] bf16 row-major
                     const float* __restrict__ XP,  // [T][32][H] fp32
                     void* __restrict__ OUT,        // !LAST: bf16 act [B][T][H]; LAST: fp32 out + h_n
                     bf16* __restrict__ hbuf,       // [2][32][H] bf16 (buf 0 pre-zeroed; unused if NWG==1)
                     unsigned* __restrict__ cnt, int T)
{
  constexpr int NP = H / NWG;                 // 128 output columns per workgroup
  extern __shared__ bf16 lds[];
  bf16* wslice = lds;                         // [NP][H] Whh row slice
  bf16* hstage = lds + (size_t)NP * H;        // NWG>1: [32][H]; NWG==1: [2][32][H]
  const int jbase = blockIdx.x * NP;
  const int tid = threadIdx.x;
  const int lane = tid & 31;
  const int lm = lane & 15, hi = lane >> 4;
  const int wave = tid >> 5;                  // 16 waves: one 16x16 output tile each
  const int mt = wave >> 3, nt = wave & 7;    // 2 M-tiles x 8 N-tiles

  { // preload Whh slice (contiguous: LDS layout == row-major slice of Whh)
    const uint4* src = (const uint4*)(Whh + (size_t)jbase * H);
    uint4* dst = (uint4*)wslice;
    for (int i = tid; i < NP * H / 8; i += (int)blockDim.x) dst[i] = src[i];
  }
  if (NWG == 1) { // h_0 = 0 in LDS buffer 0
    uint4 z = {0u, 0u, 0u, 0u};
    uint4* dst = (uint4*)hstage;
    for (int i = tid; i < 32 * H / 8; i += (int)blockDim.x) dst[i] = z;
  }
  __syncthreads();

  const int nloc = nt * 16 + lm;
  const int j = jbase + nloc;
  const bf16* bro = wslice + (size_t)nloc * H;

  for (int t = 0; t < T; ++t) {
    const bf16* hs_cur;
    bf16* hs_nxt = nullptr;    // NWG==1 path
    bf16* hnxt_g = nullptr;    // NWG>1 path
    if (NWG == 1) {
      hs_cur = hstage + (size_t)(t & 1) * 32 * H;
      hs_nxt = hstage + (size_t)((t + 1) & 1) * 32 * H;
    } else {
      // stage h_{t-1} from global into LDS once per step (one L2 round, not one per wave)
      const uint4* src = (const uint4*)(hbuf + (size_t)(t & 1) * 32 * H);
      uint4* dst = (uint4*)hstage;
      for (int i = tid; i < 32 * H / 8; i += (int)blockDim.x) dst[i] = src[i];
      __syncthreads();
      hs_cur = hstage;
      hnxt_g = hbuf + (size_t)((t + 1) & 1) * 32 * H;
    }

    // prefetch next timestep's xp slab while the WMMA chain runs
    if (t + 1 < T) {
      const float* xpn = XP + (size_t)(t + 1) * 32 * H + j;
      __builtin_prefetch(xpn + (size_t)(mt * 16 + hi * 8) * H, 0, 0);
      __builtin_prefetch(xpn + (size_t)(mt * 16 + hi * 8 + 4) * H, 0, 0);
    }

    const bf16* aro = hs_cur + (size_t)(mt * 16 + lm) * H;
    v8f acc = {};
#pragma unroll 4
    for (int k0 = 0; k0 < H; k0 += 32) {
      FragBF a, b;
      a.h[0] = *(const v8bf*)(aro + k0 + hi * 8);
      a.h[1] = *(const v8bf*)(aro + k0 + hi * 8 + 16);
      b.h[0] = *(const v8bf*)(bro + k0 + hi * 8);
      b.h[1] = *(const v8bf*)(bro + k0 + hi * 8 + 16);
      acc = wmma_bf16(a.v, b.v, acc);
    }

    const float* xp = XP + (size_t)t * 32 * H + j;
#pragma unroll
    for (int r = 0; r < 8; ++r) {
      const int b = mt * 16 + hi * 8 + r;     // C/D layout: m = r + 8*hi
      float v = acc[r] + xp[(size_t)b * H];
      v = tanhf(v);
      if (NWG == 1) hs_nxt[(size_t)b * H + j] = (bf16)v;   // ds_store_b16
      else          hnxt_g[(size_t)b * H + j] = (bf16)v;
      if (!LAST) {
        ((bf16*)OUT)[((size_t)b * T + t) * H + j] = (bf16)fmaxf(v, 0.0f);
      } else {
        ((float*)OUT)[((size_t)b * T + t) * H + j] = v;
        if (t == T - 1)
          ((float*)OUT)[(size_t)32 * T * H + (size_t)b * H + j] = v;  // h_n tail
      }
    }

    if (NWG == 1) {
      __syncthreads();  // next step reads hs_nxt
    } else {
      __syncthreads();
      if (tid == 0) {
        __threadfence();
        unsigned ticket = atomicAdd(cnt, 1u);
        unsigned target = (ticket / (unsigned)NWG + 1u) * (unsigned)NWG;
        while (__hip_atomic_load(cnt, __ATOMIC_RELAXED, __HIP_MEMORY_SCOPE_AGENT) < target)
          __builtin_amdgcn_s_sleep(1);
        __threadfence();
      }
      __syncthreads();
    }
  }
}

// ---------------------------------------------------------------- host
extern "C" void kernel_launch(void* const* d_in, const int* in_sizes, int n_in,
                              void* d_out, int out_size, void* d_ws, size_t ws_size,
                              hipStream_t stream)
{
  (void)in_sizes; (void)n_in; (void)out_size; (void)ws_size;
  const int B = 32, T = 512, M = B * T;

  const float* x    = (const float*)d_in[0];
  const float* Wih0 = (const float*)d_in[1];
  const float* Whh0 = (const float*)d_in[2];
  const float* bih0 = (const float*)d_in[3];
  const float* bhh0 = (const float*)d_in[4];
  const float* Wih1 = (const float*)d_in[5];
  const float* Whh1 = (const float*)d_in[6];
  const float* bih1 = (const float*)d_in[7];
  const float* bhh1 = (const float*)d_in[8];
  const float* Wih2 = (const float*)d_in[9];
  const float* Whh2 = (const float*)d_in[10];
  const float* bih2 = (const float*)d_in[11];
  const float* bhh2 = (const float*)d_in[12];
  float* out = (float*)d_out;

  char* p = (char*)d_ws;
  auto take = [&](size_t bytes) { char* r = p; p += (bytes + 255) & ~(size_t)255; return r; };
  bf16*  xbf   = (bf16*) take((size_t)M * 128 * 2);
  bf16*  wih0b = (bf16*) take((size_t)512 * 128 * 2);
  bf16*  whh0b = (bf16*) take((size_t)512 * 512 * 2);
  bf16*  wih1b = (bf16*) take((size_t)512 * 512 * 2);
  bf16*  whh1b = (bf16*) take((size_t)512 * 512 * 2);
  bf16*  wih2b = (bf16*) take((size_t)128 * 512 * 2);
  bf16*  whh2b = (bf16*) take((size_t)128 * 128 * 2);
  float* xp    = (float*)take((size_t)M * 512 * 4);
  bf16*  act0  = (bf16*) take((size_t)M * 512 * 2);
  bf16*  act1  = (bf16*) take((size_t)M * 512 * 2);
  bf16*  hbuf  = (bf16*) take((size_t)2 * 32 * 512 * 2);
  unsigned* cnt = (unsigned*)take(256);

  // CDNA5: up to 320KB LDS per workgroup; H=512 kernel uses 128KB weights + 32KB h stage.
  const int lds512 = (128 * 512 + 32 * 512) * 2;        // 160 KB
  const int lds128 = (128 * 128 + 2 * 32 * 128) * 2;    // 48 KB
  (void)hipFuncSetAttribute((const void*)rnn_step_kernel<512, 4, false>,
                            hipFuncAttributeMaxDynamicSharedMemorySize, lds512);
  (void)hipFuncSetAttribute((const void*)rnn_step_kernel<128, 1, true>,
                            hipFuncAttributeMaxDynamicSharedMemorySize, lds128);

  auto cvt = [&](const float* s, bf16* d, int n) {
    int blocks = (n + 255) / 256; if (blocks > 4096) blocks = 4096;
    cvt_f32_bf16<<<blocks, 256, 0, stream>>>(s, d, n);
  };
  cvt(x,    xbf,   M * 128);
  cvt(Wih0, wih0b, 512 * 128);
  cvt(Whh0, whh0b, 512 * 512);
  cvt(Wih1, wih1b, 512 * 512);
  cvt(Whh1, whh1b, 512 * 512);
  cvt(Wih2, wih2b, 128 * 512);
  cvt(Whh2, whh2b, 128 * 128);

  auto proj = [&](const bf16* X, const bf16* W, const float* bi, const float* bh,
                  float* Y, int K, int N) {
    int groups = (M / 16) * (N / 64);   // 4 N-tiles per wave
    int blocks = (groups + 7) / 8;      // 8 waves (256 threads) per block
    if (blocks > 2048) blocks = 2048;
    proj_kernel<<<blocks, 256, 0, stream>>>(X, W, bi, bh, Y, M, K, N, T);
  };

  // ---- Layer 0: 128 -> 512, ReLU
  proj(xbf, wih0b, bih0, bhh0, xp, 128, 512);
  hipMemsetAsync(hbuf, 0, (size_t)2 * 32 * 512 * 2, stream);
  hipMemsetAsync(cnt, 0, 256, stream);
  rnn_step_kernel<512, 4, false><<<4, 512, lds512, stream>>>(
      whh0b, xp, (void*)act0, hbuf, cnt, T);

  // ---- Layer 1: 512 -> 512, ReLU
  proj(act0, wih1b, bih1, bhh1, xp, 512, 512);
  hipMemsetAsync(hbuf, 0, (size_t)2 * 32 * 512 * 2, stream);
  hipMemsetAsync(cnt, 0, 256, stream);
  rnn_step_kernel<512, 4, false><<<4, 512, lds512, stream>>>(
      whh1b, xp, (void*)act1, hbuf, cnt, T);

  // ---- Layer 2: 512 -> 128, fp32 sequence + h_n, h lives entirely in LDS
  proj(act1, wih2b, bih2, bhh2, xp, 512, 128);
  rnn_step_kernel<128, 1, true><<<1, 512, lds128, stream>>>(
      whh2b, xp, (void*)out, hbuf, cnt, T);
}
